// DDiTBlockCausal_37160057045679
// MI455X (gfx1250) — compile-verified, matrix operand order checked by reading, not run on hardware
//
#include <hip/hip_runtime.h>

// ---------------------------------------------------------------------------
// Types for CDNA5 WMMA (wave32): v_wmma_f32_16x16x32_bf16
// ---------------------------------------------------------------------------
typedef __attribute__((ext_vector_type(16))) __bf16 v16bf;
typedef __attribute__((ext_vector_type(8)))  float  v8f;

union FragBF {
  v16bf v;
  uint4 u[2];
};

static __device__ __forceinline__ v8f zero8() {
  v8f z = {0.f, 0.f, 0.f, 0.f, 0.f, 0.f, 0.f, 0.f};
  return z;
}

// float -> bf16 bits, round-to-nearest-even
static __device__ __forceinline__ unsigned short f2bf(float f) {
  union { float f; unsigned int u; } c; c.f = f;
  unsigned int u = c.u;
  unsigned int r = (u + 0x7fffu + ((u >> 16) & 1u)) >> 16;
  return (unsigned short)r;
}

// Load one 16x32 bf16 A/B fragment row-slice for this lane.
// CDNA5 16-bit A layout: lane l (lo=l&15 selects row, hi=l>>4):
//   VGPRs 0..3 hold K = hi*8 .. hi*8+7, VGPRs 4..7 hold K = hi*8+16 .. +23.
static __device__ __forceinline__ v16bf load_frag_row(const unsigned short* rowp,
                                                      int k0, int hi) {
  FragBF f;
  f.u[0] = *(const uint4*)(rowp + k0 + hi * 8);
  f.u[1] = *(const uint4*)(rowp + k0 + hi * 8 + 16);
  return f.v;
}

// ---------------------------------------------------------------------------
// Elementwise f32 -> bf16 (weight conversion)
// ---------------------------------------------------------------------------
__global__ void cvt_f32_bf16(const float* __restrict__ in,
                             unsigned short* __restrict__ out, int n) {
  int i = blockIdx.x * 256 + threadIdx.x;
  if (i < n) out[i] = f2bf(in[i]);
}

// ---------------------------------------------------------------------------
// LayerNorm over last dim (1024), output bf16. One block per row.
// ---------------------------------------------------------------------------
__global__ __launch_bounds__(256) void ln_bf16_kernel(const float* __restrict__ x,
                                                      const float* __restrict__ w,
                                                      unsigned short* __restrict__ out) {
  int row = blockIdx.x;
  const float* xr = x + (size_t)row * 1024;
  __shared__ float s1[256], s2[256];
  int tid = threadIdx.x;
  float a = 0.f, b = 0.f;
  #pragma unroll
  for (int i = tid; i < 1024; i += 256) { float v = xr[i]; a += v; b += v * v; }
  s1[tid] = a; s2[tid] = b;
  __syncthreads();
  for (int st = 128; st > 0; st >>= 1) {
    if (tid < st) { s1[tid] += s1[tid + st]; s2[tid] += s2[tid + st]; }
    __syncthreads();
  }
  float mu   = s1[0] * (1.f / 1024.f);
  float var  = s2[0] * (1.f / 1024.f) - mu * mu;
  float rstd = rsqrtf(var + 1e-5f);
  for (int i = tid; i < 1024; i += 256)
    out[(size_t)row * 1024 + i] = f2bf((xr[i] - mu) * rstd * w[i]);
}

// ---------------------------------------------------------------------------
// Generic bf16 WMMA GEMM: C[M,N] = A[M,K] * W[N,K]^T  (+bias, +gelu, +residual)
// Block = 256 threads = 8 waves; wave tile 32x64; block tile 128x128.
// ---------------------------------------------------------------------------
__global__ __launch_bounds__(256) void gemm_bf16_wmma(
    const unsigned short* __restrict__ A,
    const unsigned short* __restrict__ W,
    const float* __restrict__ bias,
    const float* __restrict__ resid,
    float* __restrict__ outF,
    unsigned short* __restrict__ outB,
    int N, int K, int do_gelu) {
  int lane = threadIdx.x & 31;
  int wave = threadIdx.x >> 5;
  int lo = lane & 15, hi = lane >> 4;
  int m0 = blockIdx.x * 128 + (wave >> 1) * 32;
  int n0 = blockIdx.y * 128 + (wave & 1) * 64;

  v8f acc[2][4];
  #pragma unroll
  for (int i = 0; i < 2; i++)
    #pragma unroll
    for (int j = 0; j < 4; j++) acc[i][j] = zero8();

  const unsigned short* arow[2];
  const unsigned short* wrow[4];
  #pragma unroll
  for (int i = 0; i < 2; i++) arow[i] = A + (size_t)(m0 + 16 * i + lo) * K;
  #pragma unroll
  for (int j = 0; j < 4; j++) wrow[j] = W + (size_t)(n0 + 16 * j + lo) * K;

  for (int k0 = 0; k0 < K; k0 += 32) {
    v16bf af[2], wf[4];
    #pragma unroll
    for (int i = 0; i < 2; i++) af[i] = load_frag_row(arow[i], k0, hi);
    #pragma unroll
    for (int j = 0; j < 4; j++) wf[j] = load_frag_row(wrow[j], k0, hi);
    #pragma unroll
    for (int i = 0; i < 2; i++)
      #pragma unroll
      for (int j = 0; j < 4; j++)
        acc[i][j] = __builtin_amdgcn_wmma_f32_16x16x32_bf16(
            false, af[i], false, wf[j], (short)0, acc[i][j], false, false);
  }

  // Epilogue. C layout: lane holds col = lo; VGPR r holds row r + 8*hi.
  #pragma unroll
  for (int i = 0; i < 2; i++) {
    #pragma unroll
    for (int j = 0; j < 4; j++) {
      #pragma unroll
      for (int r = 0; r < 8; r++) {
        int row = m0 + 16 * i + r + 8 * hi;
        int col = n0 + 16 * j + lo;
        size_t off = (size_t)row * N + col;
        float v = acc[i][j][r];
        if (bias) v += bias[col];
        if (do_gelu) {
          float t = 0.7978845608f * (v + 0.044715f * v * v * v);
          v = 0.5f * v * (1.f + tanhf(t));
        }
        if (resid) v += resid[off];
        if (outF) outF[off] = v;
        else      outB[off] = f2bf(v);
      }
    }
  }
}

// ---------------------------------------------------------------------------
// RoPE + split QKV:
//   qkv f32 [b,s,3,h,64] -> q,k bf16 [b,h,s,64] (rotated), v bf16 transposed
//   [b,h,64,s] so that the PV WMMA B-fragment loads are contiguous.
// One thread per (b,s,h,i) pair, i in [0,32).
// ---------------------------------------------------------------------------
__global__ void rope_split_kernel(const float* __restrict__ qkv,
                                  const float* __restrict__ rcos,
                                  const float* __restrict__ rsin,
                                  unsigned short* __restrict__ qo,
                                  unsigned short* __restrict__ ko,
                                  unsigned short* __restrict__ vt) {
  int idx = blockIdx.x * 256 + threadIdx.x;   // B*S*H*32 = 2^21 threads
  int i = idx & 31;
  int h = (idx >> 5) & 15;
  int s = (idx >> 9) & 2047;
  int b = idx >> 20;
  size_t base = ((size_t)(b * 2048 + s)) * 3072 + h * 64;
  float c  = rcos[s * 64 + i];   // rot_cos (1,s,1,1,64); first 32 used
  float sn = rsin[s * 64 + i];
  size_t qout = ((size_t)((b * 16 + h) * 2048 + s)) * 64;
  {
    float x1 = qkv[base + i], x2 = qkv[base + i + 32];
    qo[qout + i]      = f2bf(x1 * c - x2 * sn);
    qo[qout + i + 32] = f2bf(x2 * c + x1 * sn);
  }
  {
    float x1 = qkv[base + 1024 + i], x2 = qkv[base + 1024 + i + 32];
    ko[qout + i]      = f2bf(x1 * c - x2 * sn);
    ko[qout + i + 32] = f2bf(x2 * c + x1 * sn);
  }
  {
    float x1 = qkv[base + 2048 + i], x2 = qkv[base + 2048 + i + 32];
    size_t vb = ((size_t)(b * 16 + h)) * 64 * 2048;
    vt[vb + (size_t)i * 2048 + s]        = f2bf(x1);
    vt[vb + (size_t)(i + 32) * 2048 + s] = f2bf(x2);
  }
}

// ---------------------------------------------------------------------------
// Flash attention (causal), one wave per 16-row Q tile.
// ---------------------------------------------------------------------------
static __device__ __forceinline__ float red_max16(float v) {
  v = fmaxf(v, __shfl_xor(v, 1, 32));
  v = fmaxf(v, __shfl_xor(v, 2, 32));
  v = fmaxf(v, __shfl_xor(v, 4, 32));
  v = fmaxf(v, __shfl_xor(v, 8, 32));
  return v;
}
static __device__ __forceinline__ float red_sum16(float v) {
  v += __shfl_xor(v, 1, 32);
  v += __shfl_xor(v, 2, 32);
  v += __shfl_xor(v, 4, 32);
  v += __shfl_xor(v, 8, 32);
  return v;
}

__global__ __launch_bounds__(128) void attn_kernel(
    const unsigned short* __restrict__ Q,
    const unsigned short* __restrict__ Km,
    const unsigned short* __restrict__ Vt,
    unsigned short* __restrict__ O) {
  __shared__ __align__(16) unsigned short pbuf[4][16][32];  // P tile per wave
  int lane = threadIdx.x & 31;
  int wave = threadIdx.x >> 5;
  int lo = lane & 15, hi = lane >> 4;
  int tile  = blockIdx.x * 4 + wave;     // 4096 q-tiles total
  int bh    = tile >> 7;                 // 128 tiles per (b,h)
  int qbase = (tile & 127) * 16;
  int b = bh >> 4, h = bh & 15;

  const unsigned short* Qp = Q  + (size_t)bh * 2048 * 64;
  const unsigned short* Kp = Km + (size_t)bh * 2048 * 64;
  const unsigned short* Vp = Vt + (size_t)bh * 64 * 2048;

  v16bf aq[2];
  {
    const unsigned short* rowp = Qp + (size_t)(qbase + lo) * 64;
    aq[0] = load_frag_row(rowp, 0, hi);
    aq[1] = load_frag_row(rowp, 32, hi);
  }

  v8f oacc[4];
  #pragma unroll
  for (int dt = 0; dt < 4; dt++) oacc[dt] = zero8();
  float mi[8], li[8];
  #pragma unroll
  for (int r = 0; r < 8; r++) { mi[r] = -1e30f; li[r] = 0.f; }

  int nchunks = (qbase + 47) >> 5;   // ceil((qbase+16)/32)
  for (int kc = 0; kc < nchunks; ++kc) {
    int kk0 = kc * 32;
    // S = Q * K^T over the 32-key chunk: two 16x16 score tiles
    v8f st[2];
    #pragma unroll
    for (int t = 0; t < 2; t++) {
      st[t] = zero8();
      const unsigned short* krow = Kp + (size_t)(kk0 + 16 * t + lo) * 64;
      v16bf b0 = load_frag_row(krow, 0, hi);
      v16bf b1 = load_frag_row(krow, 32, hi);
      st[t] = __builtin_amdgcn_wmma_f32_16x16x32_bf16(false, aq[0], false, b0,
                                                      (short)0, st[t], false, false);
      st[t] = __builtin_amdgcn_wmma_f32_16x16x32_bf16(false, aq[1], false, b1,
                                                      (short)0, st[t], false, false);
    }
    // Online softmax. Lane holds col = lo (+16t), VGPR r holds row r + 8*hi.
    float p0[8], p1[8], alpha[8];
    #pragma unroll
    for (int r = 0; r < 8; r++) {
      int rowg = qbase + r + 8 * hi;
      float s0 = st[0][r] * 0.125f;
      float s1 = st[1][r] * 0.125f;
      if (kk0 + lo      > rowg) s0 = -1e30f;
      if (kk0 + 16 + lo > rowg) s1 = -1e30f;
      float tm = red_max16(fmaxf(s0, s1));
      float mn = fmaxf(mi[r], tm);
      float al = __expf(mi[r] - mn);
      mi[r] = mn;
      float e0 = __expf(s0 - mn);
      float e1 = __expf(s1 - mn);
      li[r] = li[r] * al + red_sum16(e0 + e1);
      alpha[r] = al;
      p0[r] = e0; p1[r] = e1;
    }
    #pragma unroll
    for (int dt = 0; dt < 4; dt++)
      #pragma unroll
      for (int r = 0; r < 8; r++) oacc[dt][r] *= alpha[r];

    // C-layout f32 -> A-layout bf16 via LDS round trip (wave-private tile).
    #pragma unroll
    for (int r = 0; r < 8; r++) {
      pbuf[wave][r + 8 * hi][lo]      = f2bf(p0[r]);
      pbuf[wave][r + 8 * hi][16 + lo] = f2bf(p1[r]);
    }
    asm volatile("s_wait_dscnt 0" ::: "memory");
    __builtin_amdgcn_wave_barrier();
    v16bf pf;
    {
      FragBF f;
      f.u[0] = *(const uint4*)&pbuf[wave][lo][hi * 8];
      f.u[1] = *(const uint4*)&pbuf[wave][lo][hi * 8 + 16];
      pf = f.v;
    }
    asm volatile("s_wait_dscnt 0" ::: "memory");
    __builtin_amdgcn_wave_barrier();

    // O += P * V ; V^T rows are d-columns, contiguous along keys.
    #pragma unroll
    for (int dt = 0; dt < 4; dt++) {
      const unsigned short* vrow = Vp + (size_t)(16 * dt + lo) * 2048 + kk0;
      v16bf bv = load_frag_row(vrow, 0, hi);
      oacc[dt] = __builtin_amdgcn_wmma_f32_16x16x32_bf16(false, pf, false, bv,
                                                         (short)0, oacc[dt], false, false);
    }
  }

  // Normalize and store o as [b,s,D] bf16 for the out-projection GEMM.
  #pragma unroll
  for (int dt = 0; dt < 4; dt++)
    #pragma unroll
    for (int r = 0; r < 8; r++) {
      int srow = qbase + r + 8 * hi;
      float ov = oacc[dt][r] / li[r];
      O[((size_t)(b * 2048 + srow)) * 1024 + h * 64 + 16 * dt + lo] = f2bf(ov);
    }
}

// ---------------------------------------------------------------------------
// Host-side orchestration
// ---------------------------------------------------------------------------
extern "C" void kernel_launch(void* const* d_in, const int* in_sizes, int n_in,
                              void* d_out, int out_size, void* d_ws, size_t ws_size,
                              hipStream_t stream) {
  const float* x      = (const float*)d_in[0];
  const float* rcos   = (const float*)d_in[1];
  const float* rsin   = (const float*)d_in[2];
  const float* ln1_w  = (const float*)d_in[3];
  const float* w_qkv  = (const float*)d_in[4];
  const float* w_out  = (const float*)d_in[5];
  const float* ln2_w  = (const float*)d_in[6];
  const float* w_mlp1 = (const float*)d_in[7];
  const float* b_mlp1 = (const float*)d_in[8];
  const float* w_mlp2 = (const float*)d_in[9];
  const float* b_mlp2 = (const float*)d_in[10];
  float* out = (float*)d_out;

  const int M = 4096;  // B*S
  char* ws = (char*)d_ws;
  size_t off = 0;
  auto alloc = [&](size_t bytes) -> void* {
    void* p = ws + off;
    off += (bytes + 255) & ~(size_t)255;
    return p;
  };
  unsigned short* h1   = (unsigned short*)alloc((size_t)M * 1024 * 2);
  unsigned short* wqkv = (unsigned short*)alloc((size_t)3072 * 1024 * 2);
  unsigned short* wo   = (unsigned short*)alloc((size_t)1024 * 1024 * 2);
  unsigned short* w1   = (unsigned short*)alloc((size_t)4096 * 1024 * 2);
  unsigned short* w2   = (unsigned short*)alloc((size_t)1024 * 4096 * 2);
  float*          qkvf = (float*)alloc((size_t)M * 3072 * 4);
  unsigned short* qb   = (unsigned short*)alloc((size_t)M * 1024 * 2);
  unsigned short* kb   = (unsigned short*)alloc((size_t)M * 1024 * 2);
  unsigned short* vtb  = (unsigned short*)alloc((size_t)M * 1024 * 2);
  unsigned short* ob   = (unsigned short*)alloc((size_t)M * 1024 * 2);
  float*          x2   = (float*)alloc((size_t)M * 1024 * 4);
  unsigned short* h2   = (unsigned short*)alloc((size_t)M * 1024 * 2);
  unsigned short* u    = (unsigned short*)alloc((size_t)M * 4096 * 2);

  // Weights fp32 -> bf16 (cached in the 192MB L2 across all GEMM launches).
  cvt_f32_bf16<<<(3072 * 1024) / 256, 256, 0, stream>>>(w_qkv,  wqkv, 3072 * 1024);
  cvt_f32_bf16<<<(1024 * 1024) / 256, 256, 0, stream>>>(w_out,  wo,   1024 * 1024);
  cvt_f32_bf16<<<(4096 * 1024) / 256, 256, 0, stream>>>(w_mlp1, w1,   4096 * 1024);
  cvt_f32_bf16<<<(1024 * 4096) / 256, 256, 0, stream>>>(w_mlp2, w2,   1024 * 4096);

  // h1 = LN(x, ln1_w)
  ln_bf16_kernel<<<M, 256, 0, stream>>>(x, ln1_w, h1);

  // qkv = h1 @ w_qkv^T
  gemm_bf16_wmma<<<dim3(32, 24), 256, 0, stream>>>(h1, wqkv, nullptr, nullptr,
                                                   qkvf, nullptr, 3072, 1024, 0);

  // RoPE + split into per-head q,k and transposed v
  rope_split_kernel<<<(2 * 2048 * 16 * 32) / 256, 256, 0, stream>>>(
      qkvf, rcos, rsin, qb, kb, vtb);

  // causal flash attention
  attn_kernel<<<1024, 128, 0, stream>>>(qb, kb, vtb, ob);

  // x2 = x + o @ w_out^T
  gemm_bf16_wmma<<<dim3(32, 8), 256, 0, stream>>>(ob, wo, nullptr, x,
                                                  x2, nullptr, 1024, 1024, 0);

  // h2 = LN(x2, ln2_w)
  ln_bf16_kernel<<<M, 256, 0, stream>>>(x2, ln2_w, h2);

  // u = gelu(h2 @ w_mlp1^T + b_mlp1)  (bf16 out)
  gemm_bf16_wmma<<<dim3(32, 32), 256, 0, stream>>>(h2, w1, b_mlp1, nullptr,
                                                   nullptr, u, 4096, 1024, 1);

  // out = x2 + u @ w_mlp2^T + b_mlp2
  gemm_bf16_wmma<<<dim3(32, 8), 256, 0, stream>>>(u, w2, b_mlp2, x2,
                                                  out, nullptr, 1024, 4096, 0);
}
